// LinearDiscriminator_74586402063123
// MI455X (gfx1250) — matrix operand order repaired and checked by reading, not compiled, not run on hardware
//
#include <hip/hip_runtime.h>
#include <hip/hip_bf16.h>

// ---------------------------------------------------------------------------
// LinearDiscriminator: [T,B,H] -> MLP(1024->100->10->1) -> log_sigmoid ->
// masked sum over T -> exp -> [B]
// Layer-1 GEMM (M=65536,N=100,K=1024) done with v_wmma_f32_16x16x32_bf16.
// W1 fragments double-buffered in LDS via global_load_async_to_lds_b128.
// ---------------------------------------------------------------------------

typedef __attribute__((ext_vector_type(16))) __bf16 v16bf;
typedef __attribute__((ext_vector_type(8)))  float  v8f;

#define T_DIM 2048
#define B_DIM 32
#define H_DIM 1024
#define L1D   100
#define NT    7     // 7 N-tiles of 16 -> 112 (cols 100..111 zero-padded)
#define KTS   32    // 32 K-steps of 32 -> 1024
#define ROWS_PER_BLOCK 64
#define THREADS 128
#define FRAG_DWORDS 256              // one 32x16 bf16 fragment = 256 dwords
#define BATCH_DWORDS (NT * FRAG_DWORDS)   // 1792 dwords = 7168 B per k-step
#define WPACK_DWORDS (KTS * BATCH_DWORDS) // 57344 dwords = 229376 B
#define WPACK_BYTES  (WPACK_DWORDS * 4)

#if __has_builtin(__builtin_amdgcn_global_load_async_to_lds_b128)
#define HAVE_ASYNC_LDS 1
typedef __attribute__((__vector_size__(16))) int v4i;
typedef __attribute__((address_space(1))) v4i as1_v4i;   // global v4i
typedef __attribute__((address_space(3))) v4i as3_v4i;   // shared v4i
#endif

#if __has_builtin(__builtin_amdgcn_s_wait_asynccnt)
#define WAIT_ASYNC(n) __builtin_amdgcn_s_wait_asynccnt(n)
#else
#define WAIT_ASYNC(n) asm volatile("s_wait_asynccnt %0" ::"n"(n))
#endif

union AFrag { unsigned u[8]; v16bf v; };
union BFrag { unsigned u[8]; v16bf v; };

__device__ __forceinline__ unsigned bf16_of(float f) {
  unsigned x = __float_as_uint(f);
  return (x + 0x7fffu + ((x >> 16) & 1u)) >> 16;    // round-to-nearest-even
}
__device__ __forceinline__ unsigned pack_bf16(float lo, float hi) {
  return bf16_of(lo) | (bf16_of(hi) << 16);
}

// --------------------------------------------------------------------------
// Pack W1 [100,1024] f32 into WMMA B-fragment layout (bf16), N padded to 112.
// Fragment (kt,j): B[k][n] = W1[j*16 + n][kt*32 + k].
// Per-fragment layout: dword index = lane*8 + v ; lane = n + 16*((k/8)%2),
// v = 4*(k/16) + (k%8)/2, low half-word = even k.
// --------------------------------------------------------------------------
__global__ void pack_w1_kernel(const float* __restrict__ W1,
                               unsigned* __restrict__ wpack) {
  int frag = blockIdx.x;          // frag = kt*NT + j
  int kt = frag / NT, j = frag % NT;
  int tid = threadIdx.x;          // 0..255
  int lane = tid >> 3;            // 0..31
  int v    = tid & 7;             // 0..7
  int n = j * 16 + (lane & 15);
  int k = kt * 32 + ((v >> 2) << 4) + ((lane >> 4) << 3) + (v & 3) * 2;
  float f0 = 0.f, f1 = 0.f;
  if (n < L1D) {
    f0 = W1[n * H_DIM + k];
    f1 = W1[n * H_DIM + k + 1];
  }
  wpack[frag * FRAG_DWORDS + lane * 8 + v] = pack_bf16(f0, f1);
}

__global__ void zero_acc_kernel(float* acc) {
  if (threadIdx.x < B_DIM) acc[threadIdx.x] = 0.f;
}

__global__ void finalize_kernel(const float* __restrict__ acc,
                                float* __restrict__ out) {
  if (threadIdx.x < B_DIM) out[threadIdx.x] = expf(acc[threadIdx.x]);
}

// Stage one k-step's 7 B-fragments (1792 dwords) global -> LDS, async.
__device__ __forceinline__ void stage_batch(const unsigned* __restrict__ gsrc,
                                            unsigned* ldst, int tid) {
  if (tid < 112) {                         // 448 x 16B chunks, 4 per thread
#pragma unroll
    for (int i = 0; i < 4; ++i) {
      int c = i * 112 + tid;
#ifdef HAVE_ASYNC_LDS
      __builtin_amdgcn_global_load_async_to_lds_b128(
          (as1_v4i*)(gsrc + c * 4), (as3_v4i*)(ldst + c * 4), 0, 0);
#else
      *(uint4*)(ldst + c * 4) = *(const uint4*)(gsrc + c * 4);
#endif
    }
  }
}

__launch_bounds__(THREADS, 1)
__global__ void mlp_main_kernel(const float* __restrict__ X,
                                const int* __restrict__ lengths,
                                const unsigned* __restrict__ wpack,
                                const float* __restrict__ b1,
                                const float* __restrict__ W2,
                                const float* __restrict__ b2,
                                const float* __restrict__ W3,
                                const float* __restrict__ b3,
                                float* __restrict__ accOut) {
  __shared__ unsigned sB[2][BATCH_DWORDS];   // double-buffered W1 fragments
  __shared__ float    sH1[4][16][112];       // per-wave layer-1 activations
  __shared__ float    sAcc[B_DIM];           // per-block logp accumulators

  const int tid  = threadIdx.x;
  const int wave = tid >> 5;
  const int lane = tid & 31;
  const int half = lane >> 4;
  const int l16  = lane & 15;
  const int rowBase = blockIdx.x * ROWS_PER_BLOCK + wave * 16;

  if (tid < B_DIM) sAcc[tid] = 0.f;

  v8f acc[NT];
#pragma unroll
  for (int j = 0; j < NT; ++j) acc[j] = v8f{};

  // Prime batch 0
  stage_batch(wpack, &sB[0][0], tid);

  for (int kt = 0; kt < KTS; ++kt) {
    const int cur = kt & 1;
    __syncthreads();                         // buf[cur^1] free for refill
    if (kt + 1 < KTS) {
      stage_batch(wpack + (kt + 1) * BATCH_DWORDS, &sB[cur ^ 1][0], tid);
      WAIT_ASYNC(4);                         // batch kt fully landed
    } else {
      WAIT_ASYNC(0);
    }
    __syncthreads();                         // all waves' parts of buf[cur] done

    // ---- A fragment: 16 rows x 32 K, f32 -> bf16, ISA 16-bit A layout ----
    const float* rowp =
        X + (size_t)(rowBase + l16) * H_DIM + kt * 32 + half * 8;
    float4 f0 = *(const float4*)(rowp + 0);
    float4 f1 = *(const float4*)(rowp + 4);
    float4 f2 = *(const float4*)(rowp + 16);
    float4 f3 = *(const float4*)(rowp + 20);
    AFrag a;
    a.u[0] = pack_bf16(f0.x, f0.y); a.u[1] = pack_bf16(f0.z, f0.w);
    a.u[2] = pack_bf16(f1.x, f1.y); a.u[3] = pack_bf16(f1.z, f1.w);
    a.u[4] = pack_bf16(f2.x, f2.y); a.u[5] = pack_bf16(f2.z, f2.w);
    a.u[6] = pack_bf16(f3.x, f3.y); a.u[7] = pack_bf16(f3.z, f3.w);

    // ---- load all 7 B fragments first (14 ds_load_b128, one wait) ----
    BFrag b[NT];
#pragma unroll
    for (int j = 0; j < NT; ++j) {
      const uint4* p = (const uint4*)&sB[cur][j * FRAG_DWORDS + lane * 8];
      uint4 lo = p[0], hi = p[1];
      b[j].u[0] = lo.x; b[j].u[1] = lo.y; b[j].u[2] = lo.z; b[j].u[3] = lo.w;
      b[j].u[4] = hi.x; b[j].u[5] = hi.y; b[j].u[6] = hi.z; b[j].u[7] = hi.w;
    }
    // ---- then 7 back-to-back WMMAs (independent accumulators) ----
#pragma unroll
    for (int j = 0; j < NT; ++j) {
      acc[j] = __builtin_amdgcn_wmma_f32_16x16x32_bf16(
          false, a.v, false, b[j].v, (short)0, acc[j], false, false);
    }
  }

  // ---- bias + ReLU, stash h1 to LDS (C layout: lane=col, vgpr r = row) ----
#pragma unroll
  for (int j = 0; j < NT; ++j) {
    int col = j * 16 + l16;
    if (col < L1D) {
      float bias = b1[col];
#pragma unroll
      for (int r = 0; r < 8; ++r) {
        float v = acc[j][r] + bias;
        sH1[wave][r + half * 8][col] = v > 0.f ? v : 0.f;
      }
    }
  }

  // ---- tiny layers 2 & 3 + log_sigmoid + masked accumulation ----
  if (lane < 16) {
    const int row = lane;
    const int m = rowBase + row;
    float hrow[L1D];
#pragma unroll 20
    for (int k = 0; k < L1D; ++k) hrow[k] = sH1[wave][row][k];

    float h2[10];
#pragma unroll
    for (int o = 0; o < 10; ++o) {
      float s = b2[o];
      const float* w = W2 + o * L1D;
#pragma unroll 20
      for (int k = 0; k < L1D; ++k) s += w[k] * hrow[k];
      h2[o] = s > 0.f ? s : 0.f;
    }
    float z = b3[0];
#pragma unroll
    for (int o = 0; o < 10; ++o) z += W3[o] * h2[o];

    // log_sigmoid(z) = min(z,0) - log1p(exp(-|z|))
    float lp = fminf(z, 0.f) - log1pf(expf(-fabsf(z)));

    const int b = m & (B_DIM - 1);
    const int t = m >> 5;
    if (t < lengths[b]) atomicAdd(&sAcc[b], lp);   // ds_add_f32
  }
  __syncthreads();
  if (tid < B_DIM) atomicAdd(&accOut[tid], sAcc[tid]);  // global_atomic_add_f32
}

// ---------------------------------------------------------------------------
extern "C" void kernel_launch(void* const* d_in, const int* in_sizes, int n_in,
                              void* d_out, int out_size, void* d_ws,
                              size_t ws_size, hipStream_t stream) {
  const float* X       = (const float*)d_in[0];
  const int*   lengths = (const int*)d_in[1];
  const float* W1      = (const float*)d_in[2];
  const float* b1      = (const float*)d_in[3];
  const float* W2      = (const float*)d_in[4];
  const float* b2      = (const float*)d_in[5];
  const float* W3      = (const float*)d_in[6];
  const float* b3      = (const float*)d_in[7];

  unsigned* wpack = (unsigned*)d_ws;
  float*    accB  = (float*)((char*)d_ws + WPACK_BYTES);

  zero_acc_kernel<<<1, 32, 0, stream>>>(accB);
  pack_w1_kernel<<<KTS * NT, 256, 0, stream>>>(W1, wpack);
  mlp_main_kernel<<<(T_DIM * B_DIM) / ROWS_PER_BLOCK, THREADS, 0, stream>>>(
      X, lengths, wpack, b1, W2, b2, W3, b3, accB);
  finalize_kernel<<<1, 32, 0, stream>>>(accB, (float*)d_out);
}
